// TransformerTTS_59571196396011
// MI455X (gfx1250) — matrix operand order repaired and checked by reading, not compiled, not run on hardware
//
#include <hip/hip_runtime.h>

// ---------------------------------------------------------------------------
// TransformerTTS forward for MI455X (gfx1250, wave32, WMMA).
// Pure-bf16 GEMM pipeline: weights transposed+converted to bf16 once per
// launch; activations produced in bf16 by producer kernels / GEMM epilogues.
// Hot GEMM loop: async global->LDS copies (ASYNCcnt) double-buffered against
// v_wmma_f32_16x16x32_bf16 (BK=64 -> 8 WMMAs per barrier pair).
// ---------------------------------------------------------------------------

typedef __attribute__((ext_vector_type(16))) __bf16 v16bf;
typedef __attribute__((ext_vector_type(8)))  float  v8f;
typedef int v4i __attribute__((vector_size(16)));

#define LDSW 72   // fast-kernel LDS row stride (ushorts): 144B, 16B-aligned

// model dims
#define CB   8      // batch
#define CS   256    // text seq len
#define CE   512    // embed dim
#define CHF  2048   // ffn hidden
#define CL   4      // layers
#define COUT 80     // mel bins
#define CTS  1024   // spectrogram frames
#define CNH  8      // heads
#define CDH  64     // head dim
#define CTM  1536   // MAX_EXP

typedef long long ll;
typedef unsigned short u16;

#ifndef __has_builtin
#define __has_builtin(x) 0
#endif

#if __has_builtin(__builtin_amdgcn_global_load_async_to_lds_b128)
#define USE_ASYNC 1
#define ASYNC_GPTR(p) (__attribute__((address_space(1))) v4i*)(p)
#define ASYNC_LPTR(p) (__attribute__((address_space(3))) v4i*)(p)
#if __has_builtin(__builtin_amdgcn_s_wait_asynccnt)
#define WAIT_ASYNC(n) __builtin_amdgcn_s_wait_asynccnt(n)
#else
#define WAIT_ASYNC(n) asm volatile("s_wait_asynccnt %0" ::"n"(n) : "memory")
#endif
#endif

__device__ __forceinline__ u16 f2bf(float f) {
  unsigned int x = __float_as_uint(f);
  x += 0x7FFFu + ((x >> 16) & 1u);  // round-to-nearest-even
  return (u16)(x >> 16);
}

union Frag { v16bf v; uint4 q[2]; };

// ---------------------------------------------------------------------------
// fast bf16 GEMM: C = alpha*A.B (+bias) (+relu)
// A: [M][K] bf16 (lda), B: [N][K] bf16 (ldb)  -- both K-contiguous.
// OM: 0 = f32 C[gm*ldc+gn]; 1 = bf16 C[gm*ldc+gn]; 2 = bf16 C[gn*ldc+gm].
// blockIdx.z -> (zb = z/nh, zh = z%nh) with independent batch/head strides.
// Requires M%BM==0, N%BN==0, K%64==0, lda/ldb%8==0.
// ---------------------------------------------------------------------------
template <int BM, int BN, bool BIAS, bool RELU, int OM>
__global__ __launch_bounds__(256) void gemm_bf16(
    const u16* __restrict__ Ag, const u16* __restrict__ Bg,
    const float* __restrict__ bias, void* __restrict__ Cg,
    int M, int N, int K, int lda, int ldb, int ldc, int nh,
    ll sAb, ll sAh, ll sBb, ll sBh, ll sCb, ll sCh, float alpha) {
#ifdef USE_ASYNC
  __shared__ u16 As[2][BM][LDSW];
  __shared__ u16 Bs[2][BN][LDSW];
#else
  __shared__ u16 As[1][BM][LDSW];
  __shared__ u16 Bs[1][BN][LDSW];
#endif

  const int z = blockIdx.z, zb = z / nh, zh = z % nh;
  const u16* A  = Ag + zb * sAb + zh * sAh;
  const u16* Bm = Bg + zb * sBb + zh * sBh;

  const int tile_m = blockIdx.y * BM;
  const int tile_n = blockIdx.x * BN;
  const int tid  = threadIdx.x;
  const int wave = tid >> 5, lane = tid & 31;
  constexpr int WN = BN / 32;
  const int wm = (wave / WN) * 32;
  const int wn = (wave % WN) * 32;
  const int half = lane >> 4, l16 = lane & 15;

  v8f acc[2][2] = {};

  // staging mapping: item i -> row = i>>3, k8 = (i&7)*8 (8 bf16 = one uint4)
  constexpr int AIT = BM / 32;   // 16B chunks per thread for A tile (BM x 64)
  constexpr int BIT = BN / 32;
  const u16* apt[AIT];
#pragma unroll
  for (int it = 0; it < AIT; ++it) {
    int i = tid + it * 256;
    apt[it] = A + (ll)(tile_m + (i >> 3)) * lda + (i & 7) * 8;
  }
  const u16* bpt[BIT];
#pragma unroll
  for (int it = 0; it < BIT; ++it) {
    int i = tid + it * 256;
    bpt[it] = Bm + (ll)(tile_n + (i >> 3)) * ldb + (i & 7) * 8;
  }

  // ISA 7.12.2 fragment layouts (wave32):
  // A 16x32: lanes0-15 K={0..7,16..23}, lanes16-31 K={8..15,24..31}
  // B 32x16: lanes0-15 K={0..7,16..23}, lanes16-31 K={8..15,24..31}; N = lane&15
  auto mma_step = [&](int p, int ks) {
    Frag fa[2], fb[2];
#pragma unroll
    for (int mt = 0; mt < 2; ++mt) {
      int m = wm + mt * 16 + l16;
      fa[mt].q[0] = *(const uint4*)&As[p][m][ks + half * 8];
      fa[mt].q[1] = *(const uint4*)&As[p][m][ks + half * 8 + 16];
    }
#pragma unroll
    for (int nt = 0; nt < 2; ++nt) {
      int n = wn + nt * 16 + l16;
      fb[nt].q[0] = *(const uint4*)&Bs[p][n][ks + half * 8];
      fb[nt].q[1] = *(const uint4*)&Bs[p][n][ks + half * 8 + 16];
    }
#pragma unroll
    for (int mt = 0; mt < 2; ++mt)
#pragma unroll
      for (int nt = 0; nt < 2; ++nt)
        acc[mt][nt] = __builtin_amdgcn_wmma_f32_16x16x32_bf16(
            false, fa[mt].v, false, fb[nt].v, (short)0, acc[mt][nt], false, false);
  };

  const int nk = K >> 6;

#ifdef USE_ASYNC
  // gfx1250 async global->LDS copy path, double-buffered
  constexpr int NIN = AIT + BIT;  // async ops in flight per issued tile
  auto issue = [&](int p, int k0) {
#pragma unroll
    for (int it = 0; it < AIT; ++it) {
      int i = tid + it * 256;
      __builtin_amdgcn_global_load_async_to_lds_b128(
          ASYNC_GPTR(apt[it] + k0),
          ASYNC_LPTR(&As[p][i >> 3][(i & 7) * 8]), 0, 0);
    }
#pragma unroll
    for (int it = 0; it < BIT; ++it) {
      int i = tid + it * 256;
      __builtin_amdgcn_global_load_async_to_lds_b128(
          ASYNC_GPTR(bpt[it] + k0),
          ASYNC_LPTR(&Bs[p][i >> 3][(i & 7) * 8]), 0, 0);
    }
  };
  issue(0, 0);
  for (int kt = 0; kt < nk; ++kt) {
    int p = kt & 1;
    if (kt + 1 < nk) {
      issue(1 - p, (kt + 1) << 6);  // overlaps both WMMA stages of tile p
      WAIT_ASYNC(NIN);              // in-order completion -> tile p resident
    } else {
      WAIT_ASYNC(0);
    }
    __syncthreads();
    mma_step(p, 0);
    mma_step(p, 32);
    __syncthreads();
  }
#else
  // register-staged fallback
  uint4 ar[AIT], br[BIT];
  auto gload = [&](int k0) {
#pragma unroll
    for (int it = 0; it < AIT; ++it) ar[it] = *(const uint4*)(apt[it] + k0);
#pragma unroll
    for (int it = 0; it < BIT; ++it) br[it] = *(const uint4*)(bpt[it] + k0);
  };
  auto lstore = [&]() {
#pragma unroll
    for (int it = 0; it < AIT; ++it) {
      int i = tid + it * 256;
      *(uint4*)&As[0][i >> 3][(i & 7) * 8] = ar[it];
    }
#pragma unroll
    for (int it = 0; it < BIT; ++it) {
      int i = tid + it * 256;
      *(uint4*)&Bs[0][i >> 3][(i & 7) * 8] = br[it];
    }
  };
  gload(0);
  for (int kt = 0; kt < nk; ++kt) {
    lstore();
    __syncthreads();
    if (kt + 1 < nk) gload((kt + 1) << 6);
    mma_step(0, 0);
    mma_step(0, 32);
    __syncthreads();
  }
#endif

  // C layout: VGPR r -> row r (lanes0-15) / r+8 (lanes16-31); col = lane&15
#pragma unroll
  for (int mt = 0; mt < 2; ++mt)
#pragma unroll
    for (int nt = 0; nt < 2; ++nt) {
      int gn = tile_n + wn + nt * 16 + l16;
      float bv = BIAS ? bias[gn] : 0.0f;
#pragma unroll
      for (int r = 0; r < 8; ++r) {
        int gm = tile_m + wm + mt * 16 + half * 8 + r;
        float v = acc[mt][nt][r] * alpha + bv;
        if (RELU) v = fmaxf(v, 0.0f);
        if constexpr (OM == 0)
          ((float*)Cg + zb * sCb + zh * sCh)[(ll)gm * ldc + gn] = v;
        else if constexpr (OM == 1)
          ((u16*)Cg + zb * sCb + zh * sCh)[(ll)gm * ldc + gn] = f2bf(v);
        else
          ((u16*)Cg + zb * sCb + zh * sCh)[(ll)gn * ldc + gm] = f2bf(v);
      }
    }
}

// ---------------------------------------------------------------------------
// guarded fp32 GEMM (arbitrary M/N/K): C = A*B + bias. For proj/fc/dp2 only.
// ---------------------------------------------------------------------------
__global__ __launch_bounds__(256) void gemm_f32_guard(
    const float* __restrict__ A, const float* __restrict__ Bm,
    const float* __restrict__ bias, float* __restrict__ C,
    int M, int N, int K, int lda, int ldb, int ldc) {
  __shared__ u16 As[64][40];
  __shared__ u16 Bs[128][40];
  const int tile_m = blockIdx.y * 64;
  const int tile_n = blockIdx.x * 128;
  const int tid  = threadIdx.x;
  const int wave = tid >> 5, lane = tid & 31;
  const int wm = (wave >> 2) * 32;
  const int wn = (wave & 3) * 32;
  const int half = lane >> 4, l16 = lane & 15;

  v8f acc[2][2] = {};
  for (int k0 = 0; k0 < K; k0 += 32) {
    for (int i = tid; i < 64 * 32; i += 256) {
      int m = i >> 5, k = i & 31;
      int gm = tile_m + m, gk = k0 + k;
      float v = (gm < M && gk < K) ? A[(ll)gm * lda + gk] : 0.0f;
      As[m][k] = f2bf(v);
    }
    for (int i = tid; i < 128 * 32; i += 256) {
      int k = i >> 7, n = i & 127;
      int gn = tile_n + n, gk = k0 + k;
      float v = (gn < N && gk < K) ? Bm[(ll)gk * ldb + gn] : 0.0f;
      Bs[n][k] = f2bf(v);
    }
    __syncthreads();
    Frag fa[2], fb[2];
#pragma unroll
    for (int mt = 0; mt < 2; ++mt) {
      int m = wm + mt * 16 + l16;
      fa[mt].q[0] = *(const uint4*)&As[m][half * 8];
      fa[mt].q[1] = *(const uint4*)&As[m][half * 8 + 16];
    }
#pragma unroll
    for (int nt = 0; nt < 2; ++nt) {
      int n = wn + nt * 16 + l16;
      fb[nt].q[0] = *(const uint4*)&Bs[n][half * 8];
      fb[nt].q[1] = *(const uint4*)&Bs[n][half * 8 + 16];
    }
#pragma unroll
    for (int mt = 0; mt < 2; ++mt)
#pragma unroll
      for (int nt = 0; nt < 2; ++nt)
        acc[mt][nt] = __builtin_amdgcn_wmma_f32_16x16x32_bf16(
            false, fa[mt].v, false, fb[nt].v, (short)0, acc[mt][nt], false, false);
    __syncthreads();
  }
#pragma unroll
  for (int mt = 0; mt < 2; ++mt)
#pragma unroll
    for (int nt = 0; nt < 2; ++nt) {
      int gn = tile_n + wn + nt * 16 + l16;
      float bv = (gn < N) ? bias[gn] : 0.0f;
#pragma unroll
      for (int r = 0; r < 8; ++r) {
        int gm = tile_m + wm + mt * 16 + half * 8 + r;
        if (gm < M && gn < N) C[(ll)gm * ldc + gn] = acc[mt][nt][r] + bv;
      }
    }
}

// weight transpose + convert: in fp32 [K][N] -> out bf16 [N][K]; K,N % 32 == 0
__global__ __launch_bounds__(256) void wtrans_kernel(
    const float* __restrict__ in, u16* __restrict__ out, int K, int N) {
  __shared__ u16 t[32][33];
  int kt = blockIdx.y * 32, nt = blockIdx.x * 32;
  int tx = threadIdx.x & 31, ty = threadIdx.x >> 5;  // 32 x 8
#pragma unroll
  for (int r = 0; r < 32; r += 8)
    t[ty + r][tx] = f2bf(in[(ll)(kt + ty + r) * N + nt + tx]);
  __syncthreads();
#pragma unroll
  for (int r = 0; r < 32; r += 8)
    out[(ll)(nt + ty + r) * K + kt + tx] = t[tx][ty + r];
}

// flat fp32 -> bf16
__global__ __launch_bounds__(256) void cvt_kernel(
    const float* __restrict__ in, u16* __restrict__ out, int n) {
  int i = blockIdx.x * 256 + threadIdx.x;
  if (i < n) out[i] = f2bf(in[i]);
}

// row softmax: read f32 scores, write normalized bf16 probs. len % 256 == 0, len<=1536.
__global__ __launch_bounds__(256) void softmax_kernel(
    const float* __restrict__ S, u16* __restrict__ P, int len) {
  const float* p = S + (ll)blockIdx.x * len;
  u16* q = P + (ll)blockIdx.x * len;
  __shared__ float red[256];
  int tid = threadIdx.x;
  float m = -1e30f;
  for (int i = tid; i < len; i += 256) m = fmaxf(m, p[i]);
  red[tid] = m; __syncthreads();
  for (int s = 128; s > 0; s >>= 1) { if (tid < s) red[tid] = fmaxf(red[tid], red[tid + s]); __syncthreads(); }
  m = red[0]; __syncthreads();
  float ev[6];
  float sum = 0.0f;
  int cnt = len >> 8;
  for (int j = 0; j < cnt; ++j) { float e = __expf(p[tid + j * 256] - m); ev[j] = e; sum += e; }
  red[tid] = sum; __syncthreads();
  for (int s = 128; s > 0; s >>= 1) { if (tid < s) red[tid] += red[tid + s]; __syncthreads(); }
  float inv = 1.0f / red[0];
  for (int j = 0; j < cnt; ++j) q[tid + j * 256] = f2bf(ev[j] * inv);
}

// out = LayerNorm(x + r)*g + b ; dual write fp32 (residual chain) + bf16 (GEMM A)
__global__ __launch_bounds__(256) void add_ln_kernel(
    const float* x, const float* r, const float* g, const float* b,
    float* out, u16* outb) {
  ll row = blockIdx.x;
  int tid = threadIdx.x;
  const float* xp = x + row * CE;
  const float* rp = r + row * CE;
  float v0 = xp[tid] + rp[tid];
  float v1 = xp[tid + 256] + rp[tid + 256];
  __shared__ float red[256];
  red[tid] = v0 + v1; __syncthreads();
  for (int s = 128; s > 0; s >>= 1) { if (tid < s) red[tid] += red[tid + s]; __syncthreads(); }
  float mu = red[0] * (1.0f / CE); __syncthreads();
  float d0 = v0 - mu, d1 = v1 - mu;
  red[tid] = d0 * d0 + d1 * d1; __syncthreads();
  for (int s = 128; s > 0; s >>= 1) { if (tid < s) red[tid] += red[tid + s]; __syncthreads(); }
  float rstd = rsqrtf(red[0] * (1.0f / CE) + 1e-5f);
  float y0 = d0 * rstd * g[tid] + b[tid];
  float y1 = d1 * rstd * g[tid + 256] + b[tid + 256];
  out[row * CE + tid] = y0;
  out[row * CE + tid + 256] = y1;
  outb[row * CE + tid] = f2bf(y0);
  outb[row * CE + tid + 256] = f2bf(y1);
}

__global__ __launch_bounds__(256) void embed_kernel(
    const int* __restrict__ text, const float* __restrict__ emb,
    float* __restrict__ out, u16* __restrict__ outb) {
  int bs = blockIdx.x, tid = threadIdx.x;
  ll tok = text[bs];
  float v0 = emb[tok * CE + tid], v1 = emb[tok * CE + tid + 256];
  out[(ll)bs * CE + tid] = v0;
  out[(ll)bs * CE + tid + 256] = v1;
  outb[(ll)bs * CE + tid] = f2bf(v0);
  outb[(ll)bs * CE + tid + 256] = f2bf(v1);
}

// inclusive scan of durations, one block (==S threads) per batch row
__global__ __launch_bounds__(256) void cumsum_kernel(const int* __restrict__ dur, int* __restrict__ csum) {
  __shared__ int buf[CS];
  int tid = threadIdx.x, b = blockIdx.x;
  buf[tid] = dur[b * CS + tid];
  __syncthreads();
  for (int off = 1; off < CS; off <<= 1) {
    int v = (tid >= off) ? buf[tid - off] : 0;
    __syncthreads();
    buf[tid] += v;
    __syncthreads();
  }
  csum[b * CS + tid] = buf[tid];
}

// frame t of batch b copies token j = searchsorted(csum, t, 'right'); zero past total.
// Emits bf16 (reg is only consumed as a GEMM A operand).
__global__ __launch_bounds__(128) void regulate_kernel(
    const float* __restrict__ enc, const int* __restrict__ csum, u16* __restrict__ out) {
  int b = blockIdx.x / CTM, t = blockIdx.x % CTM;
  int tid = threadIdx.x;
  const int* c = csum + b * CS;
  u16* o = out + ((ll)b * CTM + t) * CE;
  if (t >= c[CS - 1]) {
    for (int i = tid; i < CE; i += 128) o[i] = 0;
    return;
  }
  int lo = 0, hi = CS;
  while (lo < hi) { int mid = (lo + hi) >> 1; if (c[mid] <= t) lo = mid + 1; else hi = mid; }
  int idx = lo < CS - 1 ? lo : CS - 1;
  const float* src = enc + ((ll)b * CS + idx) * CE;
  for (int i = tid; i < CE; i += 128) o[i] = f2bf(src[i]);
}

// ---------------------------------------------------------------------------
// host side
// ---------------------------------------------------------------------------

enum {
  IN_TEXT = 0, IN_DUR, IN_SPEC,
  IN_EMB,
  IN_ENC_WQ, IN_ENC_WK, IN_ENC_WV, IN_ENC_WO,
  IN_ENC_BQ, IN_ENC_BK, IN_ENC_BV, IN_ENC_BO,
  IN_SA_WQ, IN_SA_WK, IN_SA_WV, IN_SA_WO,
  IN_SA_BQ, IN_SA_BK, IN_SA_BV, IN_SA_BO,
  IN_CA_WQ, IN_CA_WK, IN_CA_WV, IN_CA_WO,
  IN_CA_BQ, IN_CA_BK, IN_CA_BV, IN_CA_BO,
  IN_ENC_W1, IN_ENC_B1, IN_ENC_W2, IN_ENC_B2,
  IN_DEC_W1, IN_DEC_B1, IN_DEC_W2, IN_DEC_B2,
  IN_ENC_LN1_G, IN_ENC_LN1_B, IN_ENC_LN2_G, IN_ENC_LN2_B,
  IN_DEC_LN1_G, IN_DEC_LN1_B, IN_DEC_LN2_G, IN_DEC_LN2_B, IN_DEC_LN3_G, IN_DEC_LN3_B,
  IN_DP_W1, IN_DP_B1, IN_DP_W2, IN_DP_B2,
  IN_PROJ_W, IN_PROJ_B, IN_FC_W, IN_FC_B
};

// projection-style GEMM on bf16 operands; OM selects output form
static void proj_f32(hipStream_t st, const u16* A, const u16* B, const float* bias, float* C,
                     int M, int N, int K, int lda, bool relu) {
  dim3 g(N / 128, M / 64, 1), t(256);
  if (relu)
    gemm_bf16<64, 128, true, true, 0><<<g, t, 0, st>>>(A, B, bias, C, M, N, K, lda, K, N, 1,
                                                       0, 0, 0, 0, 0, 0, 1.0f);
  else
    gemm_bf16<64, 128, true, false, 0><<<g, t, 0, st>>>(A, B, bias, C, M, N, K, lda, K, N, 1,
                                                        0, 0, 0, 0, 0, 0, 1.0f);
}
static void proj_b16(hipStream_t st, const u16* A, const u16* B, const float* bias, u16* C,
                     int M, int N, int K, int lda, bool relu) {
  dim3 g(N / 128, M / 64, 1), t(256);
  if (relu)
    gemm_bf16<64, 128, true, true, 1><<<g, t, 0, st>>>(A, B, bias, C, M, N, K, lda, K, N, 1,
                                                       0, 0, 0, 0, 0, 0, 1.0f);
  else
    gemm_bf16<64, 128, true, false, 1><<<g, t, 0, st>>>(A, B, bias, C, M, N, K, lda, K, N, 1,
                                                        0, 0, 0, 0, 0, 0, 1.0f);
}
// V projection: batched over b, writes V^T (bf16 [E][T]) per batch
static void proj_vt(hipStream_t st, const u16* A, const u16* B, const float* bias, u16* C, int T) {
  dim3 g(CE / 128, T / 64, CB), t(256);
  gemm_bf16<64, 128, true, false, 2><<<g, t, 0, st>>>(A, B, bias, C, T, CE, CE, CE, CE, T, 1,
                                                      (ll)T * CE, 0, 0, 0, (ll)CE * T, 0, 1.0f);
}
// scores = alpha * Q.K^T (both bf16, K-contig), batched (b,h) -> f32
static void gemm_scores(hipStream_t st, const u16* Q, const u16* Kt, float* Sc,
                        int M, int N, int batch, ll sAb, ll sBb, ll sCb, ll sCh) {
  dim3 g(N / 128, M / 64, batch), t(256);
  gemm_bf16<64, 128, false, false, 0><<<g, t, 0, st>>>(Q, Kt, nullptr, Sc, M, N, CDH, CE, CE, N,
                                                       CNH, sAb, CDH, sBb, CDH, sCb, sCh, 0.125f);
}
// O = P.V : P bf16 [M][K], V^T bf16 [64][K]; out bf16 strided by head
static void gemm_av(hipStream_t st, const u16* P, const u16* Vt, u16* O,
                    int M, int K, int batch, ll sAb, ll sAh, ll sBb, ll sCb) {
  dim3 g(1, M / 128, batch), t(256);
  gemm_bf16<128, 64, false, false, 1><<<g, t, 0, st>>>(P, Vt, nullptr, O, M, CDH, K, K, K, CE,
                                                       CNH, sAb, sAh, sBb, (ll)CDH * K, sCb, CDH, 1.0f);
}
static void gemm_guard(hipStream_t st, const float* A, const float* B, const float* bias, float* C,
                       int M, int N, int K, int lda, int ldb, int ldc) {
  dim3 g((N + 127) / 128, (M + 63) / 64, 1), t(256);
  gemm_f32_guard<<<g, t, 0, st>>>(A, B, bias, C, M, N, K, lda, ldb, ldc);
}

extern "C" void kernel_launch(void* const* d_in, const int* in_sizes, int n_in,
                              void* d_out, int out_size, void* d_ws, size_t ws_size,
                              hipStream_t stream) {
  auto fp = [&](int i) { return (const float*)d_in[i]; };
  const int* text = (const int*)d_in[IN_TEXT];
  const int* dur  = (const int*)d_in[IN_DUR];
  float* dout = (float*)d_out;

  // workspace carve-out (16B-aligned), ~300 MB total
  size_t off = 0;
  auto alloc = [&](size_t bytes) {
    off = (off + 15) & ~(size_t)15;
    void* p = (char*)d_ws + off;
    off += bytes;
    return p;
  };
  auto fa = [&](size_t n) { return (float*)alloc(n * 4); };
  auto ua = [&](size_t n) { return (u16*)alloc(n * 2); };

  float* enc_x  = fa((size_t)CB * CS * CE);
  float* dec_x  = fa((size_t)CB * CTS * CE);
  float* to2    = fa((size_t)CB * CTS * CE);
  float* sc     = fa((size_t)CNH * CTS * CTM);
  float* tf32   = fa((size_t)CB * CS * CHF);     // dp hidden (fp32 for guarded dp2)
  u16* enc_xb = ua((size_t)CB * CS * CE);
  u16* dec_xb = ua((size_t)CB * CTS * CE);
  u16* regb   = ua((size_t)CB * CTM * CE);
  u16* tqb    = ua((size_t)CB * CTS * CE);
  u16* tkb    = ua((size_t)CB * CTM * CE);
  u16* vt     = ua((size_t)CB * CE * CTM);       // V^T per batch
  u16* tob    = ua((size_t)CB * CTS * CE);
  u16* tfb    = ua((size_t)CB * CTS * CHF);
  u16* scb    = ua((size_t)CNH * CTS * CTM);     // bf16 softmax probs
  int* csum   = (int*)alloc((size_t)CB * CS * 4);

  // ---- one-time weight transpose+convert (fp32 [K][N] -> bf16 [N][K]) ----
  auto wt = [&](const float* W, int K, int N) {
    u16* dst = ua((size_t)K * N);
    dim3 g(N / 32, K / 32);
    wtrans_kernel<<<g, 256, 0, stream>>>(W, dst, K, N);
    return (const u16*)dst;
  };
  const u16 *ewq[CL], *ewk[CL], *ewv[CL], *ewo[CL], *ew1[CL], *ew2[CL];
  const u16 *swq[CL], *swk[CL], *swv[CL], *swo[CL];
  const u16 *cwq[CL], *cwk[CL], *cwv[CL], *cwo[CL], *dw1[CL], *dw2[CL];
  for (int i = 0; i < CL; ++i) {
    ewq[i] = wt(fp(IN_ENC_WQ) + (size_t)i * CE * CE, CE, CE);
    ewk[i] = wt(fp(IN_ENC_WK) + (size_t)i * CE * CE, CE, CE);
    ewv[i] = wt(fp(IN_ENC_WV) + (size_t)i * CE * CE, CE, CE);
    ewo[i] = wt(fp(IN_ENC_WO) + (size_t)i * CE * CE, CE, CE);
    ew1[i] = wt(fp(IN_ENC_W1) + (size_t)i * CE * CHF, CE, CHF);
    ew2[i] = wt(fp(IN_ENC_W2) + (size_t)i * CHF * CE, CHF, CE);
    swq[i] = wt(fp(IN_SA_WQ) + (size_t)i * CE * CE, CE, CE);
    swk[i] = wt(fp(IN_SA_WK) + (size_t)i * CE * CE, CE, CE);
    swv[i] = wt(fp(IN_SA_WV) + (size_t)i * CE * CE, CE, CE);
    swo[i] = wt(fp(IN_SA_WO) + (size_t)i * CE * CE, CE, CE);
    cwq[i] = wt(fp(IN_CA_WQ) + (size_t)i * CE * CE, CE, CE);
    cwk[i] = wt(fp(IN_CA_WK) + (size_t)i * CE * CE, CE, CE);
    cwv[i] = wt(fp(IN_CA_WV) + (size_t)i * CE * CE, CE, CE);
    cwo[i] = wt(fp(IN_CA_WO) + (size_t)i * CE * CE, CE, CE);
    dw1[i] = wt(fp(IN_DEC_W1) + (size_t)i * CE * CHF, CE, CHF);
    dw2[i] = wt(fp(IN_DEC_W2) + (size_t)i * CHF * CE, CHF, CE);
  }
  const u16* dpw1T = wt(fp(IN_DP_W1), CE, CHF);

  // 1) embedding (fp32 + bf16)
  embed_kernel<<<CB * CS, 256, 0, stream>>>(text, fp(IN_EMB), enc_x, enc_xb);

  // 2) encoder layers (post-norm)
  for (int i = 0; i < CL; ++i) {
    int M = CB * CS;
    proj_b16(stream, enc_xb, ewq[i], fp(IN_ENC_BQ) + (size_t)i * CE, tqb, M, CE, CE, CE, false);
    proj_b16(stream, enc_xb, ewk[i], fp(IN_ENC_BK) + (size_t)i * CE, tkb, M, CE, CE, CE, false);
    proj_vt(stream, enc_xb, ewv[i], fp(IN_ENC_BV) + (size_t)i * CE, vt, CS);
    gemm_scores(stream, tqb, tkb, sc, CS, CS, CB * CNH,
                (ll)CS * CE, (ll)CS * CE, (ll)CNH * CS * CS, (ll)CS * CS);
    softmax_kernel<<<CB * CNH * CS, 256, 0, stream>>>(sc, scb, CS);
    gemm_av(stream, scb, vt, tob, CS, CS, CB * CNH,
            (ll)CNH * CS * CS, (ll)CS * CS, (ll)CE * CS, (ll)CS * CE);
    proj_f32(stream, tob, ewo[i], fp(IN_ENC_BO) + (size_t)i * CE, to2, M, CE, CE, CE, false);
    add_ln_kernel<<<M, 256, 0, stream>>>(enc_x, to2, fp(IN_ENC_LN1_G) + (size_t)i * CE,
                                         fp(IN_ENC_LN1_B) + (size_t)i * CE, enc_x, enc_xb);
    proj_b16(stream, enc_xb, ew1[i], fp(IN_ENC_B1) + (size_t)i * CHF, tfb, M, CHF, CE, CE, true);
    proj_f32(stream, tfb, ew2[i], fp(IN_ENC_B2) + (size_t)i * CE, to2, M, CE, CHF, CHF, false);
    add_ln_kernel<<<M, 256, 0, stream>>>(enc_x, to2, fp(IN_ENC_LN2_G) + (size_t)i * CE,
                                         fp(IN_ENC_LN2_B) + (size_t)i * CE, enc_x, enc_xb);
  }

  // 3) duration predictor -> dur_pred appended after mel output
  proj_f32(stream, enc_xb, dpw1T, fp(IN_DP_B1), tf32, CB * CS, CHF, CE, CE, true);
  gemm_guard(stream, tf32, fp(IN_DP_W2), fp(IN_DP_B2), dout + (size_t)CB * CTS * COUT,
             CB * CS, 1, CHF, CHF, 1, 1);

  // 4) length regulate (bf16 memory for decoder K/V projections)
  cumsum_kernel<<<CB, CS, 0, stream>>>(dur, csum);
  regulate_kernel<<<CB * CTM, 128, 0, stream>>>(enc_x, csum, regb);

  // 5) target projection (K=80: guarded fp32), then bf16 copy
  gemm_guard(stream, fp(IN_SPEC), fp(IN_PROJ_W), fp(IN_PROJ_B), dec_x, CB * CTS, CE, COUT, COUT, CE, CE);
  cvt_kernel<<<(CB * CTS * CE) / 256, 256, 0, stream>>>(dec_x, dec_xb, CB * CTS * CE);

  // 6) decoder layers
  for (int i = 0; i < CL; ++i) {
    int M = CB * CTS;
    // ---- self attention
    proj_b16(stream, dec_xb, swq[i], fp(IN_SA_BQ) + (size_t)i * CE, tqb, M, CE, CE, CE, false);
    proj_b16(stream, dec_xb, swk[i], fp(IN_SA_BK) + (size_t)i * CE, tkb, M, CE, CE, CE, false);
    proj_vt(stream, dec_xb, swv[i], fp(IN_SA_BV) + (size_t)i * CE, vt, CTS);
    for (int b = 0; b < CB; ++b) {  // 50MB score chunk per batch
      gemm_scores(stream, tqb + (size_t)b * CTS * CE, tkb + (size_t)b * CTS * CE, sc,
                  CTS, CTS, CNH, 0, 0, 0, (ll)CTS * CTS);
      softmax_kernel<<<CNH * CTS, 256, 0, stream>>>(sc, scb, CTS);
      gemm_av(stream, scb, vt + (size_t)b * CE * CTS, tob + (size_t)b * CTS * CE,
              CTS, CTS, CNH, 0, (ll)CTS * CTS, 0, 0);
    }
    proj_f32(stream, tob, swo[i], fp(IN_SA_BO) + (size_t)i * CE, to2, M, CE, CE, CE, false);
    add_ln_kernel<<<M, 256, 0, stream>>>(dec_x, to2, fp(IN_DEC_LN1_G) + (size_t)i * CE,
                                         fp(IN_DEC_LN1_B) + (size_t)i * CE, dec_x, dec_xb);
    // ---- cross attention (K/V from length-regulated memory)
    proj_b16(stream, dec_xb, cwq[i], fp(IN_CA_BQ) + (size_t)i * CE, tqb, M, CE, CE, CE, false);
    proj_b16(stream, regb, cwk[i], fp(IN_CA_BK) + (size_t)i * CE, tkb, CB * CTM, CE, CE, CE, false);
    proj_vt(stream, regb, cwv[i], fp(IN_CA_BV) + (size_t)i * CE, vt, CTM);
    for (int b = 0; b < CB; ++b) {
      gemm_scores(stream, tqb + (size_t)b * CTS * CE, tkb + (size_t)b * CTM * CE, sc,
                  CTS, CTM, CNH, 0, 0, 0, (ll)CTS * CTM);
      softmax_kernel<<<CNH * CTS, 256, 0, stream>>>(sc, scb, CTM);
      gemm_av(stream, scb, vt + (size_t)b * CE * CTM, tob + (size_t)b * CTS * CE,
              CTS, CTM, CNH, 0, (ll)CTS * CTM, 0, 0);
    }
    proj_f32(stream, tob, cwo[i], fp(IN_CA_BO) + (size_t)i * CE, to2, M, CE, CE, CE, false);
    add_ln_kernel<<<M, 256, 0, stream>>>(dec_x, to2, fp(IN_DEC_LN2_G) + (size_t)i * CE,
                                         fp(IN_DEC_LN2_B) + (size_t)i * CE, dec_x, dec_xb);
    // ---- ffn
    proj_b16(stream, dec_xb, dw1[i], fp(IN_DEC_B1) + (size_t)i * CHF, tfb, M, CHF, CE, CE, true);
    proj_f32(stream, tfb, dw2[i], fp(IN_DEC_B2) + (size_t)i * CE, to2, M, CE, CHF, CHF, false);
    add_ln_kernel<<<M, 256, 0, stream>>>(dec_x, to2, fp(IN_DEC_LN3_G) + (size_t)i * CE,
                                         fp(IN_DEC_LN3_B) + (size_t)i * CE, dec_x, dec_xb);
  }

  // 7) final projection -> mel output (N=80: guarded fp32)
  gemm_guard(stream, dec_x, fp(IN_FC_W), fp(IN_FC_B), dout, CB * CTS, COUT, CE, CE, COUT, COUT);
}
